// HeisenbergAttention_42245298323793
// MI455X (gfx1250) — matrix operand order repaired and checked by reading, not compiled, not run on hardware
//
#include <hip/hip_runtime.h>

#define DD 1024
#define TT 2048
#define BB 8
#define NROWS (BB * TT)  // 16384

typedef __attribute__((ext_vector_type(16))) __bf16 v16bf;
typedef __attribute__((ext_vector_type(8)))  float  v8f;
typedef __attribute__((ext_vector_type(4)))  unsigned int u32x4;
typedef __attribute__((ext_vector_type(4)))  unsigned int ui32x4;
typedef __attribute__((ext_vector_type(8)))  int i32x8;
typedef __attribute__((ext_vector_type(4)))  int i32x4;

union FragBF {
  v16bf v;
  u32x4 q[2];
  unsigned short s[16];
};

__device__ __forceinline__ unsigned short f2bf(float f) {
  unsigned u = __float_as_uint(f);
  u += 0x7fffu + ((u >> 16) & 1u);   // round-to-nearest-even
  return (unsigned short)(u >> 16);
}

// ---------------------------------------------------------------------------
// TDM: issue a 2-D tensor_load_to_lds (bf16 elements, data_size=2B).
// D# per CDNA5 ISA ch.8: group0 = {count/type, lds_addr, global_addr},
// group1 = {flags/pads, tensor dims, tile dims, dim0 stride}. Groups 2/3
// zero (2-D tile). pad_enable inserts (pad_amount+1)*4 bytes into the LDS
// destination every 2^(pad_interval+3) bytes (bank-conflict avoidance).
// ---------------------------------------------------------------------------
__device__ __forceinline__ void tdm_load_2d(unsigned lds_addr,
                                            const void* gaddr,
                                            unsigned tensor_d0, unsigned tensor_d1,
                                            unsigned tile_d0, unsigned tile_d1,
                                            unsigned long d0_stride_elems,
                                            unsigned pad_interval_code,
                                            unsigned pad_amount_code,
                                            int pad_en) {
  unsigned long ga = (unsigned long)gaddr;
  ui32x4 g0;
  g0[0] = 1u;                                            // count=1, user mode
  g0[1] = lds_addr;                                      // LDS byte address
  g0[2] = (unsigned)(ga & 0xffffffffu);                  // global_addr[31:0]
  g0[3] = (unsigned)((ga >> 32) & 0x1ffffffu) | (2u << 30);  // [56:32] | type=2

  i32x8 g1;
  unsigned w0 = (1u << 16)                               // data_size = 1 -> 2 bytes
              | ((pad_en ? 1u : 0u) << 20)
              | ((pad_interval_code & 7u) << 22)
              | ((pad_amount_code & 127u) << 25);
  g1[0] = (int)w0;
  g1[1] = (int)((tensor_d0 & 0xffffu) << 16);            // atomic_bar=0 | dim0 lo
  g1[2] = (int)(((tensor_d0 >> 16) & 0xffffu) | ((tensor_d1 & 0xffffu) << 16));
  g1[3] = (int)(((tensor_d1 >> 16) & 0xffffu) | ((tile_d0 & 0xffffu) << 16));
  g1[4] = (int)(tile_d1 & 0xffffu);                      // tile_dim1 | tile_dim2=0
  g1[5] = (int)(unsigned)(d0_stride_elems & 0xffffffffu);
  g1[6] = (int)(unsigned)((d0_stride_elems >> 32) & 0xffffu);  // dim1_stride = 0
  g1[7] = 0;

  i32x4 z4 = {0, 0, 0, 0};
#if __clang_major__ >= 23
  i32x8 z8 = {0, 0, 0, 0, 0, 0, 0, 0};
  __builtin_amdgcn_tensor_load_to_lds(g0, g1, z4, z4, z8, 0);
#else
  __builtin_amdgcn_tensor_load_to_lds(g0, g1, z4, z4, 0);
#endif
}

// ---------------------------------------------------------------------------
// Stage 1: fp32 -> bf16 conversion of x and the three weight matrices
// ---------------------------------------------------------------------------
__global__ void convert_kernel(const float* __restrict__ x,
                               const float* __restrict__ Wq,
                               const float* __restrict__ Wk,
                               const float* __restrict__ Wv,
                               unsigned short* __restrict__ xb,
                               unsigned short* __restrict__ wqb,
                               unsigned short* __restrict__ wkb,
                               unsigned short* __restrict__ wvb) {
  long idx = (long)blockIdx.x * blockDim.x + threadIdx.x;
  const long nx = (long)NROWS * DD;
  const long nw = (long)DD * DD;
  if (idx < nx) { xb[idx] = f2bf(x[idx]); return; }
  idx -= nx;
  if (idx < nw) { wqb[idx] = f2bf(Wq[idx]); return; }
  idx -= nw;
  if (idx < nw) { wkb[idx] = f2bf(Wk[idx]); return; }
  idx -= nw;
  if (idx < nw) { wvb[idx] = f2bf(Wv[idx]); }
}

// ---------------------------------------------------------------------------
// Stage 2: QKV projection GEMM (y = x @ W^T), one wave per 16x16 tile.
// z = 0:Q (also accumulates sum-of-squares per row), 1:K, 2:V (stored
// transposed per batch: vbT[b][d][t] so PV B-fragments are contiguous).
// ---------------------------------------------------------------------------
__global__ void __launch_bounds__(256) qkv_gemm_kernel(
    const unsigned short* __restrict__ xb,
    const unsigned short* __restrict__ wqb,
    const unsigned short* __restrict__ wkb,
    const unsigned short* __restrict__ wvb,
    unsigned short* __restrict__ qb,
    unsigned short* __restrict__ kb,
    unsigned short* __restrict__ vbT,
    float* __restrict__ sumsq) {
  const int lane = threadIdx.x & 31;
  const int wave = threadIdx.x >> 5;
  const int col0 = (blockIdx.x * 8 + wave) * 16;
  const int row0 = blockIdx.y * 16;
  const int z = blockIdx.z;
  const unsigned short* W = (z == 0) ? wqb : (z == 1) ? wkb : wvb;

  const int hi = lane >> 4;
  const int lr = lane & 15;

  const unsigned short* arow = xb + (long)(row0 + lr) * DD;
  const unsigned short* brow = W  + (long)(col0 + lr) * DD;

  v8f acc = {};
  for (int kk = 0; kk < DD; kk += 32) {
    FragBF a, b;
    a.q[0] = *(const u32x4*)(arow + kk + hi * 8);
    a.q[1] = *(const u32x4*)(arow + kk + hi * 8 + 16);
    b.q[0] = *(const u32x4*)(brow + kk + hi * 8);
    b.q[1] = *(const u32x4*)(brow + kk + hi * 8 + 16);
    acc = __builtin_amdgcn_wmma_f32_16x16x32_bf16(false, a.v, false, b.v,
                                                  (short)0, acc, false, false);
  }

  if (z == 0) {
    for (int i = 0; i < 8; ++i) {
      int r = row0 + i + hi * 8;
      atomicAdd(&sumsq[r], acc[i] * acc[i]);
    }
  }
  if (z == 2) {
    for (int i = 0; i < 8; ++i) {
      int r = row0 + i + hi * 8;
      int bidx = r >> 11;
      int t = r & (TT - 1);
      vbT[(long)bidx * DD * TT + (long)(col0 + lr) * TT + t] = f2bf(acc[i]);
    }
  } else {
    unsigned short* outp = (z == 0) ? qb : kb;
    for (int i = 0; i < 8; ++i) {
      int r = row0 + i + hi * 8;
      outp[(long)r * DD + col0 + lr] = f2bf(acc[i]);
    }
  }
}

// ---------------------------------------------------------------------------
// Stage 3: adaptive temperature from q-norm
// ---------------------------------------------------------------------------
__global__ void temp_kernel(const float* __restrict__ sumsq,
                            const float* __restrict__ hbar,
                            float* __restrict__ temp) {
  int r = blockIdx.x * blockDim.x + threadIdx.x;
  if (r < NROWS) {
    float t = hbar[0] / (sqrtf(sumsq[r]) + 1e-8f);
    temp[r] = fminf(fmaxf(t, 0.1f), 5.0f);
  }
}

// ---------------------------------------------------------------------------
// Stage 4: flash attention with double-buffered TDM-staged K/V tiles.
// Block = 16 query rows x 8 waves; each wave owns a 128-wide d-chunk of O.
// Pipeline: wave 0 issues the TDM for key block j+1 into the ping-pong
// buffer at the top of iteration j, computes block j from the other buffer,
// and only waits TENSORcnt==0 at the iteration tail -> DMA overlaps the
// whole S / softmax / PV pipeline of the current block.
//
// LDS per buffer:
//   K tile: 32 rows, 2048B data + 16B pad every 1024B -> 2080B stride (66560B)
//   V tile: 1024 rows, 64B data + 16B pad -> 80B stride              (81920B)
// Two buffers: 296960B dynamic (+ ~3.3KB static) < 320KB per WGP.
// ---------------------------------------------------------------------------
#define K_ROW_B   2080
#define K_LDS_B   (32 * K_ROW_B)         // 66560
#define V_ROW_B   80
#define V_LDS_B   (1024 * V_ROW_B)       // 81920
#define BUF_B     (K_LDS_B + V_LDS_B)    // 148480
#define ATTN_DYN_LDS (2 * BUF_B)         // 296960

extern __shared__ char dyn_lds[];

__device__ __forceinline__ void issue_kv_tdm(unsigned ldsbase,
                                             const unsigned short* kb,
                                             const unsigned short* vbT,
                                             int b, int key0) {
  // K: 32 rows x 1024 bf16, contiguous rows; pad 16B per 1024B
  tdm_load_2d(ldsbase, kb + ((long)b * TT + key0) * DD,
              /*tensor_d0=*/DD, /*tensor_d1=*/32,
              /*tile_d0=*/DD, /*tile_d1=*/32,
              /*d0_stride=*/DD,
              /*pad_interval=*/7, /*pad_amount=*/3, /*pad_en=*/1);
  // V: 1024 rows x 32 bf16 from vbT (row stride TT); pad 16B per 64B
  tdm_load_2d(ldsbase + K_LDS_B, vbT + (long)b * DD * TT + key0,
              /*tensor_d0=*/32, /*tensor_d1=*/DD,
              /*tile_d0=*/32, /*tile_d1=*/DD,
              /*d0_stride=*/TT,
              /*pad_interval=*/3, /*pad_amount=*/3, /*pad_en=*/1);
}

__global__ void __launch_bounds__(256) attn_kernel(
    const unsigned short* __restrict__ qb,
    const unsigned short* __restrict__ kb,
    const unsigned short* __restrict__ vbT,
    const float* __restrict__ temp,
    float* __restrict__ out) {
  __shared__ float sS[16 * 32];
  __shared__ unsigned short sP[16 * 32];
  __shared__ float sM[16], sL[16], sScale[16];

  const int lane = threadIdx.x & 31;
  const int wave = threadIdx.x >> 5;
  const int hi = lane >> 4;
  const int lr = lane & 15;
  const int b = blockIdx.y;
  const int qrow = blockIdx.x * 16;
  const long rowg = (long)b * TT + qrow;
  const int dbase = wave * 128;

  const unsigned ldsbase = (unsigned)(unsigned long)(const void*)dyn_lds;

  // Prologue: start DMA of key block 0 into buffer 0 before anything else.
  if (wave == 0) issue_kv_tdm(ldsbase, kb, vbT, b, 0);

  if (threadIdx.x < 16) {
    sM[threadIdx.x] = -1e30f;
    sL[threadIdx.x] = 0.0f;
  }
  float tempR = 0.0f;
  if (threadIdx.x < 16) tempR = temp[rowg + threadIdx.x];

  // Preload this wave's Q fragments (16 rows x 128 d) - overlaps the DMA.
  FragBF qf[4];
  {
    const unsigned short* qrowp = qb + (rowg + lr) * DD + dbase;
    for (int ks = 0; ks < 4; ++ks) {
      qf[ks].q[0] = *(const u32x4*)(qrowp + ks * 32 + hi * 16);
      qf[ks].q[1] = *(const u32x4*)(qrowp + ks * 32 + hi * 16 + 8);
    }
  }

  v8f o[8] = {};

  if (wave == 0) __builtin_amdgcn_s_wait_tensorcnt((short)0);
  __syncthreads();   // buffer 0 published

  for (int j = 0; j < TT / 32; ++j) {
    const char* sK = dyn_lds + (j & 1) * BUF_B;
    const char* sV = sK + K_LDS_B;

    // Prefetch next key block into the other buffer (overlaps compute below).
    if (wave == 0 && j + 1 < TT / 32) {
      issue_kv_tdm(ldsbase + ((j + 1) & 1) * BUF_B, kb, vbT, b, (j + 1) * 32);
    }

    // zero score staging buffer
    for (int i = threadIdx.x; i < 16 * 32; i += 256) sS[i] = 0.0f;
    __syncthreads();

    // partial S = Q K^T over this wave's 128-wide d-chunk (two 16-key tiles)
    v8f s0 = {}, s1 = {};
    for (int ks = 0; ks < 4; ++ks) {
      const int doff = dbase + ks * 32 + hi * 16;
      const int dbyte = doff * 2 + ((doff >= 512) ? 16 : 0);
      FragBF b0, b1;
      b0.q[0] = *(const u32x4*)(sK + (lr) * K_ROW_B + dbyte);
      b0.q[1] = *(const u32x4*)(sK + (lr) * K_ROW_B + dbyte + 16);
      b1.q[0] = *(const u32x4*)(sK + (16 + lr) * K_ROW_B + dbyte);
      b1.q[1] = *(const u32x4*)(sK + (16 + lr) * K_ROW_B + dbyte + 16);
      s0 = __builtin_amdgcn_wmma_f32_16x16x32_bf16(false, qf[ks].v, false, b0.v,
                                                   (short)0, s0, false, false);
      s1 = __builtin_amdgcn_wmma_f32_16x16x32_bf16(false, qf[ks].v, false, b1.v,
                                                   (short)0, s1, false, false);
    }
    for (int i = 0; i < 8; ++i) {
      int row = i + hi * 8;
      atomicAdd(&sS[row * 32 + lr], s0[i]);
      atomicAdd(&sS[row * 32 + 16 + lr], s1[i]);
    }
    __syncthreads();

    // online softmax update, one thread per query row
    if (threadIdx.x < 16) {
      int r = threadIdx.x;
      float mOld = sM[r];
      float mNew = mOld;
      float sc[32];
      for (int c = 0; c < 32; ++c) {
        float v = sS[r * 32 + c] * tempR;
        sc[c] = v;
        mNew = fmaxf(mNew, v);
      }
      float scale = __expf(mOld - mNew);
      float lNew = sL[r] * scale;
      for (int c = 0; c < 32; ++c) {
        float p = __expf(sc[c] - mNew);
        lNew += p;
        sP[r * 32 + c] = f2bf(p);
      }
      sM[r] = mNew;
      sL[r] = lNew;
      sScale[r] = scale;
    }
    __syncthreads();

    // rescale O and accumulate P @ V over this wave's d-chunk
    FragBF pf;  // A-frag of P (16 rows x 32 keys) from LDS
    pf.q[0] = *(const u32x4*)(&sP[lr * 32 + hi * 8]);
    pf.q[1] = *(const u32x4*)(&sP[lr * 32 + hi * 8 + 16]);
    float rs[8];
    for (int i = 0; i < 8; ++i) rs[i] = sScale[i + hi * 8];
    for (int t = 0; t < 8; ++t) {
      for (int i = 0; i < 8; ++i) o[t][i] *= rs[i];
      const int vrow = dbase + t * 16 + lr;
      FragBF vf;
      vf.q[0] = *(const u32x4*)(sV + vrow * V_ROW_B + hi * 16);
      vf.q[1] = *(const u32x4*)(sV + vrow * V_ROW_B + hi * 16 + 32);
      o[t] = __builtin_amdgcn_wmma_f32_16x16x32_bf16(false, pf.v, false, vf.v,
                                                     (short)0, o[t], false, false);
    }

    // next buffer must be fully landed before anyone computes on it
    if (wave == 0) __builtin_amdgcn_s_wait_tensorcnt((short)0);
    __syncthreads();
  }

  // normalize by row sums and store fp32 output
  float linv[8];
  for (int i = 0; i < 8; ++i) linv[i] = 1.0f / sL[i + hi * 8];
  for (int t = 0; t < 8; ++t) {
    int col = dbase + t * 16 + lr;
    for (int i = 0; i < 8; ++i) {
      long row = rowg + i + hi * 8;
      out[row * DD + col] = o[t][i] * linv[i];
    }
  }
}

// ---------------------------------------------------------------------------
// Launch
// ---------------------------------------------------------------------------
extern "C" void kernel_launch(void* const* d_in, const int* in_sizes, int n_in,
                              void* d_out, int out_size, void* d_ws, size_t ws_size,
                              hipStream_t stream) {
  (void)in_sizes; (void)n_in; (void)out_size; (void)ws_size;
  const float* x    = (const float*)d_in[0];
  const float* Wq   = (const float*)d_in[1];
  const float* Wk   = (const float*)d_in[2];
  const float* Wv   = (const float*)d_in[3];
  const float* hbar = (const float*)d_in[4];
  float* out = (float*)d_out;

  char* ws = (char*)d_ws;
  size_t off = 0;
  unsigned short* xb  = (unsigned short*)(ws + off); off += (size_t)NROWS * DD * 2;
  unsigned short* qb  = (unsigned short*)(ws + off); off += (size_t)NROWS * DD * 2;
  unsigned short* kb  = (unsigned short*)(ws + off); off += (size_t)NROWS * DD * 2;
  unsigned short* vbT = (unsigned short*)(ws + off); off += (size_t)NROWS * DD * 2;
  unsigned short* wqb = (unsigned short*)(ws + off); off += (size_t)DD * DD * 2;
  unsigned short* wkb = (unsigned short*)(ws + off); off += (size_t)DD * DD * 2;
  unsigned short* wvb = (unsigned short*)(ws + off); off += (size_t)DD * DD * 2;
  float* sumsq = (float*)(ws + off); off += (size_t)NROWS * 4;
  float* temp  = (float*)(ws + off); off += (size_t)NROWS * 4;

  hipMemsetAsync(sumsq, 0, (size_t)NROWS * 4, stream);

  long total = (long)NROWS * DD + 3L * DD * DD;
  convert_kernel<<<dim3((unsigned)((total + 255) / 256)), dim3(256), 0, stream>>>(
      x, Wq, Wk, Wv, xb, wqb, wkb, wvb);

  qkv_gemm_kernel<<<dim3(DD / 16 / 8, NROWS / 16, 3), dim3(256), 0, stream>>>(
      xb, wqb, wkb, wvb, qb, kb, vbT, sumsq);

  temp_kernel<<<dim3(NROWS / 256), dim3(256), 0, stream>>>(sumsq, hbar, temp);

  attn_kernel<<<dim3(TT / 16, BB), dim3(256), ATTN_DYN_LDS, stream>>>(
      qb, kb, vbT, temp, out);
}
